// CausalSelfAttention_16312285790349
// MI455X (gfx1250) — matrix operand order repaired and checked by reading, not compiled, not run on hardware
//
#include <hip/hip_runtime.h>
#include <hip/hip_bf16.h>
#include <stdint.h>

typedef unsigned short u16;
typedef unsigned int   u32;

typedef __bf16 v16bf __attribute__((ext_vector_type(16)));
typedef float  v8f   __attribute__((ext_vector_type(8)));

union Frag { u32 u[8]; v16bf bf; };
union V4U  { uint4 v; u32 w[4]; u16 h[8]; };

__device__ __forceinline__ u16 f2bf(float f) {
    u32 u = __builtin_bit_cast(u32, f);
    u32 r = u + 0x7FFFu + ((u >> 16) & 1u);   // round-to-nearest-even
    return (u16)(r >> 16);
}

// A-operand K pattern (16x32 bf16): VGPR r holds K pair starting at:
__device__ __forceinline__ int kpatA(int r, int lane) {
    return ((r >> 2) << 4) + ((lane & 16) ? 8 : 0) + ((r & 3) << 1);
}
// B-operand K pattern (32x16 bf16): same as A pattern
__device__ __forceinline__ int kpatB(int r, int lane) {
    return ((r >> 2) << 4) + ((lane & 16) ? 8 : 0) + ((r & 3) << 1);
}

// ---- DPP16 butterfly reduction within each 16-lane half (pure VALU, no LDS) ----
template <int CTRL>
__device__ __forceinline__ float dpp_bfly(float v) {
    int y = __builtin_amdgcn_update_dpp(0, __builtin_bit_cast(int, v),
                                        CTRL, 0xF, 0xF, true);
    return __builtin_bit_cast(float, y);
}
__device__ __forceinline__ float hmax16(float v) {
    v = fmaxf(v, dpp_bfly<0xB1>(v));    // quad_perm(1,0,3,2) : xor 1
    v = fmaxf(v, dpp_bfly<0x4E>(v));    // quad_perm(2,3,0,1) : xor 2
    v = fmaxf(v, dpp_bfly<0x141>(v));   // row_half_mirror    : join quads
    v = fmaxf(v, dpp_bfly<0x140>(v));   // row_mirror         : join octets
    return v;
}
__device__ __forceinline__ float hsum16(float v) {
    v += dpp_bfly<0xB1>(v);
    v += dpp_bfly<0x4E>(v);
    v += dpp_bfly<0x141>(v);
    v += dpp_bfly<0x140>(v);
    return v;
}

// ---------------- fp32 -> bf16 pack (x4 vectorized) ----------------
__global__ __launch_bounds__(256)
void cvt_bf16_kernel(const float* __restrict__ in, u16* __restrict__ out, int n4) {
    int i = blockIdx.x * 256 + threadIdx.x;
    if (i >= n4) return;
    float4 f = ((const float4*)in)[i];
    uint2 o;
    o.x = (u32)f2bf(f.x) | ((u32)f2bf(f.y) << 16);
    o.y = (u32)f2bf(f.z) | ((u32)f2bf(f.w) << 16);
    ((uint2*)out)[i] = o;
}

// ---------------- V transpose: qkv[B*T][3C] -> Vt[B][H][64][T] ----------------
__global__ __launch_bounds__(256)
void vtrans_kernel(const u16* __restrict__ qkv, u16* __restrict__ vt,
                   int B, int T, int C, int H) {
    __shared__ __align__(16) u16 tile[64][72];
    const int t0 = blockIdx.x * 64;
    const int h  = blockIdx.y;
    const int b  = blockIdx.z;
    const int ld = 3 * C;
    const u16* vp = qkv + (size_t)b * T * ld + 2 * C + (size_t)h * 64;

    {
        int r = threadIdx.x >> 2;
        int c = (threadIdx.x & 3) * 16;
        const uint4* g = (const uint4*)(vp + (size_t)(t0 + r) * ld + c);
        uint4 x0 = g[0], x1 = g[1];
        *(uint4*)(&tile[r][c])     = x0;
        *(uint4*)(&tile[r][c + 8]) = x1;
    }
    __syncthreads();
    {
        int d  = threadIdx.x >> 2;
        int ts = (threadIdx.x & 3) * 16;
        V4U o0, o1;
#pragma unroll
        for (int i = 0; i < 8; ++i) o0.h[i] = tile[ts + i][d];
#pragma unroll
        for (int i = 0; i < 8; ++i) o1.h[i] = tile[ts + 8 + i][d];
        u16* dst = vt + ((size_t)(b * H + h) * 64 + d) * T + t0 + ts;
        *(uint4*)(dst)     = o0.v;
        *(uint4*)(dst + 8) = o1.v;
    }
}

// ---------------- bf16 GEMM: C[M,N] = A[M,K] * B[K,N] + bias ----------------
// Block tile 128x256, 8 waves as 2(m) x 4(n), each wave 64x64 = 4x4 WMMA tiles.
// Double-buffered LDS staging: one barrier per 32-deep K step.
template <bool OUT_BF16>
__global__ __launch_bounds__(256)
void gemm_bf16_kernel(const u16* __restrict__ A, const u16* __restrict__ Bm,
                      const float* __restrict__ bias,
                      u16* __restrict__ Cbf, float* __restrict__ Cf,
                      int M, int N, int K) {
    constexpr int LDT = 40;                          // padded stride (80B)
    __shared__ __align__(16) u16 Asb[2][128 * LDT];  // [row][k]
    __shared__ __align__(16) u16 Bsb[2][256 * LDT];  // [col][k] (transposed)

    const int tid  = threadIdx.x;
    const int lane = tid & 31;
    const int wid  = tid >> 5;
    const int bm = blockIdx.y * 128;
    const int bn = blockIdx.x * 256;
    const int wm = (wid >> 2) * 64;   // 0,64
    const int wn = (wid & 3) * 64;    // 0,64,128,192

    const v8f zero8 = {0, 0, 0, 0, 0, 0, 0, 0};
    v8f acc[4][4];
#pragma unroll
    for (int i = 0; i < 4; ++i)
#pragma unroll
        for (int j = 0; j < 4; ++j) acc[i][j] = zero8;

    const int ar = tid >> 1;          // A: row in tile, 2 threads/row (16 elems each)
    const int ak = (tid & 1) * 16;
    const int br = tid >> 3;          // B: k-row, 8 threads/row (32 elems each)
    const int bc = (tid & 7) * 32;
    const int row = lane & 15;

    uint4 a0, a1;
    u32 bw[16];

    auto loadG = [&](int k0) {
        const uint4* ga = (const uint4*)(A + (size_t)(bm + ar) * K + k0 + ak);
        a0 = ga[0];
        a1 = ga[1];
        const uint4* gb = (const uint4*)(Bm + (size_t)(k0 + br) * N + bn + bc);
        uint4 b0 = gb[0], b1 = gb[1], b2 = gb[2], b3 = gb[3];
        *(uint4*)&bw[0]  = b0;
        *(uint4*)&bw[4]  = b1;
        *(uint4*)&bw[8]  = b2;
        *(uint4*)&bw[12] = b3;
    };
    auto storeL = [&](int buf) {
        *(uint4*)(&Asb[buf][ar * LDT + ak])     = a0;
        *(uint4*)(&Asb[buf][ar * LDT + ak + 8]) = a1;
#pragma unroll
        for (int j = 0; j < 16; ++j) {
            Bsb[buf][(bc + 2 * j)     * LDT + br] = (u16)(bw[j] & 0xFFFFu);
            Bsb[buf][(bc + 2 * j + 1) * LDT + br] = (u16)(bw[j] >> 16);
        }
    };

    loadG(0);
    storeL(0);
    __syncthreads();

    for (int k0 = 0; k0 < K; k0 += 32) {
        const int  buf = (k0 >> 5) & 1;
        const bool pf  = (k0 + 32) < K;
        if (pf) loadG(k0 + 32);              // overlap next tile's global loads

        Frag af[4];
#pragma unroll
        for (int tm = 0; tm < 4; ++tm) {
            const u16* p = &Asb[buf][(wm + tm * 16 + row) * LDT];
#pragma unroll
            for (int r = 0; r < 8; ++r) af[tm].u[r] = *(const u32*)(p + kpatA(r, lane));
        }
#pragma unroll
        for (int tn = 0; tn < 4; ++tn) {
            Frag bf;
            const u16* p = &Bsb[buf][(wn + tn * 16 + row) * LDT];
#pragma unroll
            for (int r = 0; r < 8; ++r) bf.u[r] = *(const u32*)(p + kpatB(r, lane));
#pragma unroll
            for (int tm = 0; tm < 4; ++tm)
                acc[tm][tn] = __builtin_amdgcn_wmma_f32_16x16x32_bf16(
                    false, af[tm].bf, false, bf.bf, (short)0, acc[tm][tn], false, false);
        }

        if (pf) storeL(buf ^ 1);             // fill the other buffer
        __syncthreads();
    }

    // epilogue: C layout -> VGPR v: row = v + 8*(lane>=16), col = lane&15
    const int rhalf = (lane >> 4) << 3;
    const int ncol  = lane & 15;
#pragma unroll
    for (int tm = 0; tm < 4; ++tm) {
#pragma unroll
        for (int tn = 0; tn < 4; ++tn) {
            int gn = bn + wn + tn * 16 + ncol;
            float bv = bias[gn];
#pragma unroll
            for (int v = 0; v < 8; ++v) {
                int gm = bm + wm + tm * 16 + v + rhalf;
                float val = acc[tm][tn][v] + bv;
                if (OUT_BF16) Cbf[(size_t)gm * N + gn] = f2bf(val);
                else          Cf[(size_t)gm * N + gn]  = val;
            }
        }
    }
}

// ---------------- causal flash attention ----------------
// qkv bf16 [B*T][3C]; Vt bf16 [B][H][64][T]; out bf16 [B*T][C].
// One wave per 16-query tile; branchless mask-free main loop over full 32-key
// tiles with register double-buffered K fragments, plus one masked diagonal step.
__global__ __launch_bounds__(128)
void attn_kernel(const u16* __restrict__ qkv, const u16* __restrict__ vt,
                 u16* __restrict__ out, int B, int T, int C, int H) {
    const int   hs     = 64;
    // faithful bug: scale = C // H**-0.5 = 4096; fold log2(e) for exp2-domain softmax
    const float scale2 = 4096.0f * 1.4426950408889634f;
    const int   ld     = 3 * C;

    const int lane = threadIdx.x & 31;
    const int wid  = threadIdx.x >> 5;
    const int qt   = blockIdx.x * 4 + wid;
    const int h    = blockIdx.y;
    const int b    = blockIdx.z;
    const int q_base = qt * 16;

    const u16* qp  = qkv + (size_t)b * T * ld + (size_t)h * hs;
    const u16* kp  = qp + C;
    const u16* vtp = vt + (size_t)(b * H + h) * 64 * T;

    __shared__ __align__(16) u16 Pb[4][16 * 40];
    u16* P = Pb[wid];

    const int row   = lane & 15;
    const int rhalf = (lane >> 4) << 3;

    // Q fragments (A-operand, 16x64 = two 16x32)
    Frag qf[2];
    {
        const u16* qrow = qp + (size_t)(q_base + row) * ld;
#pragma unroll
        for (int f = 0; f < 2; ++f)
#pragma unroll
            for (int r = 0; r < 8; ++r)
                qf[f].u[r] = *(const u32*)(qrow + f * 32 + kpatA(r, lane));
    }

    const v8f zero8 = {0, 0, 0, 0, 0, 0, 0, 0};
    v8f o[4];
#pragma unroll
    for (int g = 0; g < 4; ++g) o[g] = zero8;
    float mst[8], lst[8];
#pragma unroll
    for (int v = 0; v < 8; ++v) { mst[v] = -1e30f; lst[v] = 0.0f; }

    // K fragments for one 32-key tile: [sub(0,1)][dhalf(0,1)] flattened
    Frag kfA[4], kfB[4];
    auto loadK = [&](Frag* kf, int kb) {
#pragma unroll
        for (int sub = 0; sub < 2; ++sub) {
            const u16* krow = kp + (size_t)(kb + sub * 16 + row) * ld;
#pragma unroll
            for (int f = 0; f < 2; ++f)
#pragma unroll
                for (int r = 0; r < 8; ++r)
                    kf[sub * 2 + f].u[r] = *(const u32*)(krow + f * 32 + kpatB(r, lane));
        }
    };

    // one 32-key step; `masked`/`t1` are compile-time constant at each call site
    auto step = [&](const Frag* kf, int kb, bool masked, bool t1) {
        v8f s0 = zero8, s1 = zero8;
#pragma unroll
        for (int f = 0; f < 2; ++f)
            s0 = __builtin_amdgcn_wmma_f32_16x16x32_bf16(
                false, qf[f].bf, false, kf[f].bf, (short)0, s0, false, false);
        if (t1) {
#pragma unroll
            for (int f = 0; f < 2; ++f)
                s1 = __builtin_amdgcn_wmma_f32_16x16x32_bf16(
                    false, qf[f].bf, false, kf[2 + f].bf, (short)0, s1, false, false);
        }

        float alpha[8];
#pragma unroll
        for (int v = 0; v < 8; ++v) {
            float a = s0[v] * scale2;
            float c = t1 ? s1[v] * scale2 : -1e30f;
            if (masked) {
                int qi = q_base + v + rhalf;
                if (kb + row > qi)      a = -1e30f;
                if (kb + 16 + row > qi) c = -1e30f;
            }
            float rx = hmax16(fmaxf(a, c));
            float mn = fmaxf(mst[v], rx);
            alpha[v] = exp2f(mst[v] - mn);
            mst[v] = mn;
            float p0 = exp2f(a - mn);
            float p1 = exp2f(c - mn);
            s0[v] = p0; s1[v] = p1;
            lst[v] = lst[v] * alpha[v] + hsum16(p0 + p1);
        }
#pragma unroll
        for (int g = 0; g < 4; ++g)
#pragma unroll
            for (int v = 0; v < 8; ++v) o[g][v] *= alpha[v];

        // P (C-layout, fp32) -> LDS bf16 -> reload as A-fragment (16x32)
#pragma unroll
        for (int v = 0; v < 8; ++v) {
            int mr = v + rhalf;
            P[mr * 40 + row]      = f2bf(s0[v]);
            P[mr * 40 + 16 + row] = f2bf(s1[v]);
        }
        asm volatile("s_wait_dscnt 0x0" ::: "memory");
        Frag pf;
        const u16* prow = P + row * 40;
#pragma unroll
        for (int r = 0; r < 8; ++r)
            pf.u[r] = *(const u32*)(prow + kpatA(r, lane));

        // O += P * V : V B-fragments from Vt (keys contiguous -> b32/b128 loads)
#pragma unroll
        for (int g = 0; g < 4; ++g) {
            Frag vf;
            const u16* vrow = vtp + (size_t)(g * 16 + row) * T + kb;
#pragma unroll
            for (int r = 0; r < 8; ++r)
                vf.u[r] = *(const u32*)(vrow + kpatB(r, lane));
            o[g] = __builtin_amdgcn_wmma_f32_16x16x32_bf16(
                false, pf.bf, false, vf.bf, (short)0, o[g], false, false);
        }
    };

    const int q_hi     = q_base + 15;
    const int full_end = q_base & ~31;   // keys below this are never masked

    // main loop: mask-free, K-fragment register double buffering (static ping-pong)
    if (full_end > 0) {
        loadK(kfA, 0);
        int kb = 0;
        while (true) {
            if (kb + 32 < full_end) loadK(kfB, kb + 32);
            step(kfA, kb, false, true);
            kb += 32;
            if (kb >= full_end) break;
            if (kb + 32 < full_end) loadK(kfA, kb + 32);
            step(kfB, kb, false, true);
            kb += 32;
            if (kb >= full_end) break;
        }
    }
    // diagonal remainder: keys [full_end, q_hi], masked
    {
        const int kb = full_end;
        loadK(kfA, kb);
        if (kb + 16 <= q_hi) step(kfA, kb, true, true);
        else                 step(kfA, kb, true, false);
    }

    // normalize + store to [B*T][C] bf16 (head-interleaved layout for proj GEMM)
    float inv[8];
#pragma unroll
    for (int v = 0; v < 8; ++v) inv[v] = 1.0f / lst[v];
    u16* ob = out + (size_t)b * T * C + (size_t)h * hs;
#pragma unroll
    for (int g = 0; g < 4; ++g)
#pragma unroll
        for (int v = 0; v < 8; ++v) {
            int qi = q_base + v + rhalf;
            ob[(size_t)qi * C + g * 16 + row] = f2bf(o[g][v] * inv[v]);
        }
}

extern "C" void kernel_launch(void* const* d_in, const int* in_sizes, int n_in,
                              void* d_out, int out_size, void* d_ws, size_t ws_size,
                              hipStream_t stream) {
    (void)in_sizes; (void)n_in; (void)out_size; (void)ws_size;
    const float* x      = (const float*)d_in[0];
    const float* w_attn = (const float*)d_in[1];
    const float* b_attn = (const float*)d_in[2];
    const float* w_proj = (const float*)d_in[3];
    const float* b_proj = (const float*)d_in[4];
    float* out = (float*)d_out;

    const int B = 2, T = 2048, C = 1024, H = 16;
    const int M = B * T;        // 4096
    const int N3 = 3 * C;       // 3072

    // workspace layout (bf16)
    u16* xb   = (u16*)d_ws;                       // [M][C]
    u16* wab  = xb   + (size_t)M * C;             // [C][3C]
    u16* wpb  = wab  + (size_t)C * N3;            // [C][C]
    u16* qkvb = wpb  + (size_t)C * C;             // [M][3C]
    u16* aob  = qkvb + (size_t)M * N3;            // [M][C]
    u16* vtb  = aob  + (size_t)M * C;             // [B][H][64][T]

    // fp32 -> bf16 packs
    {
        int n4;
        n4 = (M * C) / 4;
        cvt_bf16_kernel<<<(n4 + 255) / 256, 256, 0, stream>>>(x, xb, n4);
        n4 = (C * N3) / 4;
        cvt_bf16_kernel<<<(n4 + 255) / 256, 256, 0, stream>>>(w_attn, wab, n4);
        n4 = (C * C) / 4;
        cvt_bf16_kernel<<<(n4 + 255) / 256, 256, 0, stream>>>(w_proj, wpb, n4);
    }

    // qkv = x @ w_attn + b_attn  -> bf16
    {
        dim3 grid(N3 / 256, M / 128);
        gemm_bf16_kernel<true><<<grid, 256, 0, stream>>>(xb, wab, b_attn, qkvb,
                                                         nullptr, M, N3, C);
    }

    // V transpose for key-contiguous B fragments
    {
        dim3 grid(T / 64, H, B);
        vtrans_kernel<<<grid, 256, 0, stream>>>(qkvb, vtb, B, T, C, H);
    }

    // causal attention -> bf16 [M][C]
    {
        dim3 grid((T / 16) / 4, H, B);
        attn_kernel<<<grid, 128, 0, stream>>>(qkvb, vtb, aob, B, T, C, H);
    }

    // out = attn_out @ w_proj + b_proj -> fp32
    {
        dim3 grid(C / 256, M / 128);
        gemm_bf16_kernel<false><<<grid, 256, 0, stream>>>(aob, wpb, b_proj, nullptr,
                                                          out, M, C, C);
    }
}
